// MultipleHeadAttention_37821482008823
// MI455X (gfx1250) — matrix operand order repaired and checked by reading, not compiled, not run on hardware
//
#include <hip/hip_runtime.h>

// ---------------- problem constants ----------------
constexpr int B_     = 8;
constexpr int S_     = 1024;
constexpr int EMB_   = 1024;
constexpr int HEADS_ = 16;
constexpr int DEPTH_ = 64;
constexpr int MROWS_ = B_ * S_;          // 8192
constexpr int LDSTR  = 40;               // padded LDS row stride (bf16 elems)

typedef __attribute__((ext_vector_type(16))) __bf16 v16bf;
typedef __attribute__((ext_vector_type(8)))  __bf16 bf16x8;
typedef __attribute__((ext_vector_type(8)))  float  v8f;

// softmax scale folded into exp2: (1/sqrt(64)) * log2(e)
constexpr float SC_LOG2E = 0.125f * 1.4426950408889634f;

__device__ inline v8f wmma_bf16(v16bf a, v16bf b, v8f c) {
  return __builtin_amdgcn_wmma_f32_16x16x32_bf16(
      /*neg_a=*/false, a, /*neg_b=*/false, b,
      /*c_mod=*/(short)0, c, /*reuse_a=*/false, /*reuse_b=*/false);
}

// max with a DPP16-permuted copy (reduction stays within each 16-lane row)
template <int CTRL>
__device__ inline float dpp_fmax(float x) {
  int xi = __builtin_bit_cast(int, x);
  int yi = __builtin_amdgcn_update_dpp(xi, xi, CTRL, 0xf, 0xf, true);
  return fmaxf(x, __builtin_bit_cast(float, yi));
}
// full 16-lane row max: xor1, xor2 (quad perms), lane^7, lane^15
__device__ inline float rowmax16(float x) {
  x = dpp_fmax<0x0B1>(x);   // quad_perm [1,0,3,2]
  x = dpp_fmax<0x04E>(x);   // quad_perm [2,3,0,1]
  x = dpp_fmax<0x141>(x);   // row_half_mirror
  x = dpp_fmax<0x140>(x);   // row_mirror
  return x;
}

// A-matrix fragment, 16(M) x 32(K) bf16, row-major source with stride (elems).
// ISA layout: lane[0..15] M=lane, K = 0-7,16-23 ; lane[16..31] M=lane-16, K = 8-15,24-31
__device__ inline v16bf load_frag_a(const __bf16* __restrict__ base, int stride, int lane) {
  const int m  = lane & 15;
  const int kb = (lane >> 4) << 3;               // 0 or 8
  const __bf16* p = base + m * stride + kb;
  bf16x8 lo = *reinterpret_cast<const bf16x8*>(p);
  bf16x8 hi = *reinterpret_cast<const bf16x8*>(p + 16);
  v16bf a;
#pragma unroll
  for (int i = 0; i < 8; ++i) { a[i] = lo[i]; a[i + 8] = hi[i]; }
  return a;
}

// B-matrix fragment, 32(K) x 16(N) bf16; source stored N-major: base[n*stride + k]
// ISA layout: lane[0..15] N=lane, K = 0-7,16-23 ; lane[16..31] N=lane-16, K = 8-15,24-31
__device__ inline v16bf load_frag_b(const __bf16* __restrict__ base, int stride, int lane) {
  const int n  = lane & 15;
  const int kb = (lane >> 4) << 3;               // 0 or 8
  const __bf16* p = base + n * stride + kb;
  bf16x8 lo = *reinterpret_cast<const bf16x8*>(p);
  bf16x8 hi = *reinterpret_cast<const bf16x8*>(p + 16);
  v16bf b;
#pragma unroll
  for (int i = 0; i < 8; ++i) { b[i] = lo[i]; b[i + 8] = hi[i]; }
  return b;
}

__device__ inline v8f zero_v8f() { v8f z = {0.f,0.f,0.f,0.f,0.f,0.f,0.f,0.f}; return z; }

// ============================================================================
// Kernel 1: per-head projection GEMM, two heads per block.
// out[h] = X[8192,1024] @ W[h][1024,64] + b[h]
// Block tile 128(M) x 128(N = 2 heads), K step 32.
// 8 waves in 4(M) x 2(N) grid; each wave a 32x64 tile = 2x4 WMMA accumulators.
// TRANS=false: out layout [B,H,S,D] (bf16) ; TRANS=true: [B,H,D,S] (bf16, for V)
// ============================================================================
template <bool TRANS>
__global__ __launch_bounds__(256)
void proj_kernel(const float* __restrict__ X,      // [MROWS, EMB]
                 const float* __restrict__ W,      // [HEADS, EMB, DEPTH]
                 const float* __restrict__ bias,   // [HEADS, DEPTH]
                 __bf16* __restrict__ out) {
  __shared__ __bf16 As[128][LDSTR];   // A tile: [m][k]
  __shared__ __bf16 Bs[128][LDSTR];   // B tile transposed: [n][k]

  const int t    = threadIdx.x;
  const int lane = t & 31;
  const int w    = t >> 5;            // 0..7
  const int wm   = w >> 1;            // 0..3 (M)
  const int wn   = w & 1;             // 0..1 (N)
  const int gm0  = blockIdx.x * 128;
  const int h0   = blockIdx.y * 2;    // first head of this block

  v8f acc[2][4];
#pragma unroll
  for (int i = 0; i < 2; ++i)
#pragma unroll
    for (int j = 0; j < 4; ++j) acc[i][j] = zero_v8f();

  const int arow = t >> 1;            // 0..127
  const int acb  = (t & 1) << 4;      // 0 or 16
  const int bk   = t >> 3;            // 0..31
  const int bnb  = (t & 7) << 4;      // 0,16,...,112 (16-aligned: never crosses head)
  const int bh_  = h0 + (bnb >> 6);
  const int be0  = bnb & 63;

  for (int k0 = 0; k0 < EMB_; k0 += 32) {
    __syncthreads();
    // stage A (f32 -> bf16): 128x32
    {
      const float* src = X + (size_t)(gm0 + arow) * EMB_ + k0 + acb;
      if (k0 + 32 < EMB_) __builtin_prefetch(src + 32, 0, 1);
#pragma unroll
      for (int i = 0; i < 16; i += 4) {
        float4 v = *reinterpret_cast<const float4*>(src + i);
        As[arow][acb + i + 0] = (__bf16)v.x;
        As[arow][acb + i + 1] = (__bf16)v.y;
        As[arow][acb + i + 2] = (__bf16)v.z;
        As[arow][acb + i + 3] = (__bf16)v.w;
      }
    }
    // stage B transposed (f32 -> bf16): W[bh_][k0+bk][be0..be0+15] -> Bs[n][k]
    {
      const float* src = W + ((size_t)bh_ * EMB_ + (k0 + bk)) * DEPTH_ + be0;
#pragma unroll
      for (int i = 0; i < 16; i += 4) {
        float4 v = *reinterpret_cast<const float4*>(src + i);
        Bs[bnb + i + 0][bk] = (__bf16)v.x;
        Bs[bnb + i + 1][bk] = (__bf16)v.y;
        Bs[bnb + i + 2][bk] = (__bf16)v.z;
        Bs[bnb + i + 3][bk] = (__bf16)v.w;
      }
    }
    __syncthreads();

    v16bf a0 = load_frag_a(&As[wm * 32 +  0][0], LDSTR, lane);
    v16bf a1 = load_frag_a(&As[wm * 32 + 16][0], LDSTR, lane);
#pragma unroll
    for (int bi = 0; bi < 4; ++bi) {
      v16bf b = load_frag_b(&Bs[wn * 64 + bi * 16][0], LDSTR, lane);
      acc[0][bi] = wmma_bf16(a0, b, acc[0][bi]);
      acc[1][bi] = wmma_bf16(a1, b, acc[1][bi]);
    }
  }

  // epilogue: C layout -> lane n = lane&15 ; row m = (lane>>4)*8 + r
  const int nlane = lane & 15;
  const int mgrp  = (lane >> 4) << 3;
#pragma unroll
  for (int ai = 0; ai < 2; ++ai) {
#pragma unroll
    for (int bi = 0; bi < 4; ++bi) {
      const int n  = wn * 64 + bi * 16 + nlane;   // 0..127 within block
      const int hh = h0 + (n >> 6);
      const int e  = n & 63;
      const float bvv = bias[hh * DEPTH_ + e];
#pragma unroll
      for (int r = 0; r < 8; ++r) {
        const int gm = gm0 + wm * 32 + ai * 16 + mgrp + r;
        const int bb = gm >> 10;        // batch
        const int ss = gm & 1023;       // seq
        const size_t head = (size_t)bb * HEADS_ + hh;
        const size_t idx = TRANS ? (head * DEPTH_ + e) * S_ + ss
                                 : (head * S_ + ss) * DEPTH_ + e;
        out[idx] = (__bf16)(acc[ai][bi][r] + bvv);
      }
    }
  }
}

// ============================================================================
// Kernel 2: flash attention per (b,h).  Each wave owns 16 query rows.
// Block = 8 waves = 128 q rows.  Streams K/V in 32-key chunks.
// Row max via DPP16 (no LDS); denominator via ones-column WMMA.
// qh,kh: [B,H,S,D] bf16 ; vT: [B,H,D,S] bf16 ; ctx out: [B,S,EMB] bf16
// ============================================================================
__global__ __launch_bounds__(256)
void attn_kernel(const __bf16* __restrict__ qh,
                 const __bf16* __restrict__ kh,
                 const __bf16* __restrict__ vT,
                 __bf16* __restrict__ ctx) {
  __shared__ __bf16 Plds[8][16][LDSTR];     // per-wave P transpose buffer

  const int t    = threadIdx.x;
  const int lane = t & 31;
  const int w    = t >> 5;
  const int bh   = blockIdx.y;
  const int bb   = bh >> 4;
  const int h    = bh & 15;
  const int qb   = blockIdx.x * 128 + w * 16;

  const __bf16* qhb = qh + (size_t)bh * S_ * DEPTH_;
  const __bf16* khb = kh + (size_t)bh * S_ * DEPTH_;
  const __bf16* vtb = vT + (size_t)bh * DEPTH_ * S_;
  __bf16 (*Pw)[LDSTR] = Plds[w];

  // preload Q fragments (16 rows x 64 depth -> 2 A-frags of 16x32)
  v16bf aq0 = load_frag_a(qhb + (size_t)qb * DEPTH_ +  0, DEPTH_, lane);
  v16bf aq1 = load_frag_a(qhb + (size_t)qb * DEPTH_ + 32, DEPTH_, lane);

  // all-ones B fragment: extra accumulator column carries the softmax sum
  v16bf ones;
#pragma unroll
  for (int i = 0; i < 16; ++i) ones[i] = (__bf16)1.0f;

  float rm[8];
  v8f   o[5];                                // o[0..3] = ctx, o[4] = running denom
#pragma unroll
  for (int r = 0; r < 8; ++r) rm[r] = -3.0e38f;
#pragma unroll
  for (int ng = 0; ng < 5; ++ng) o[ng] = zero_v8f();

  for (int sk0 = 0; sk0 < S_; sk0 += 32) {
    // ---- logits: [16q x 32k] = Q[16x64] @ K^T[64x32] ----
    v8f l0 = zero_v8f(), l1 = zero_v8f();
    {
      v16bf bk00 = load_frag_b(khb + (size_t)(sk0 +  0) * DEPTH_ +  0, DEPTH_, lane);
      v16bf bk01 = load_frag_b(khb + (size_t)(sk0 +  0) * DEPTH_ + 32, DEPTH_, lane);
      v16bf bk10 = load_frag_b(khb + (size_t)(sk0 + 16) * DEPTH_ +  0, DEPTH_, lane);
      v16bf bk11 = load_frag_b(khb + (size_t)(sk0 + 16) * DEPTH_ + 32, DEPTH_, lane);
      l0 = wmma_bf16(aq0, bk00, l0);
      l0 = wmma_bf16(aq1, bk01, l0);
      l1 = wmma_bf16(aq0, bk10, l1);
      l1 = wmma_bf16(aq1, bk11, l1);
    }
    // ---- online softmax: row max via DPP within each 16-lane half ----
    const int nlane = lane & 15;
#pragma unroll
    for (int r = 0; r < 8; ++r) {
      const float mx   = rowmax16(fmaxf(l0[r], l1[r]));
      const float mnew = fmaxf(rm[r], mx);
      const float cfac = exp2f((rm[r] - mnew) * SC_LOG2E);
      const float p0   = exp2f((l0[r] - mnew) * SC_LOG2E);
      const float p1   = exp2f((l1[r] - mnew) * SC_LOG2E);
      rm[r] = mnew;
#pragma unroll
      for (int ng = 0; ng < 5; ++ng) o[ng][r] *= cfac;   // rescales denom too
      const int m = ((lane >> 4) << 3) + r;   // C-layout row
      Pw[m][nlane]      = (__bf16)p0;
      Pw[m][16 + nlane] = (__bf16)p1;
    }
    // wave-private LDS transpose: wait for DS stores before A-frag reads
    asm volatile("s_wait_dscnt 0" ::: "memory");
    v16bf ap = load_frag_a(&Pw[0][0], LDSTR, lane);
    // ---- O += P[16x32] @ V[32x64]  ;  denom += P @ ones ----
#pragma unroll
    for (int ng = 0; ng < 4; ++ng) {
      v16bf bv = load_frag_b(vtb + (size_t)(ng * 16) * S_ + sk0, S_, lane);
      o[ng] = wmma_bf16(ap, bv, o[ng]);
    }
    o[4] = wmma_bf16(ap, ones, o[4]);
  }

  // normalize and store context in [B, S, H*D] layout
  const int nlane = lane & 15;
  const int mgrp  = (lane >> 4) << 3;
#pragma unroll
  for (int r = 0; r < 8; ++r) {
    const float inv = 1.0f / o[4][r];
    const int srow = qb + mgrp + r;
#pragma unroll
    for (int ng = 0; ng < 4; ++ng) {
      const int d = ng * 16 + nlane;
      ctx[((size_t)bb * S_ + srow) * EMB_ + h * DEPTH_ + d] = (__bf16)(o[ng][r] * inv);
    }
  }
}

// ============================================================================
// Kernel 3: output projection.  out = ctx[8192,1024] @ Wo[1024,1024] + bo (f32)
// Block tile 128x128, wave tile 32x64 (2x4 accumulators).
// ============================================================================
__global__ __launch_bounds__(256)
void outproj_kernel(const __bf16* __restrict__ ctx,  // [MROWS, EMB] bf16
                    const float* __restrict__ Wo,    // [EMB, EMB]
                    const float* __restrict__ bo,    // [EMB]
                    float* __restrict__ out) {       // [MROWS, EMB]
  __shared__ __bf16 As[128][LDSTR];
  __shared__ __bf16 Bs[128][LDSTR];

  const int t    = threadIdx.x;
  const int lane = t & 31;
  const int w    = t >> 5;
  const int wm   = w >> 1;
  const int wn   = w & 1;
  const int gm0  = blockIdx.x * 128;
  const int n0   = blockIdx.y * 128;

  v8f acc[2][4];
#pragma unroll
  for (int i = 0; i < 2; ++i)
#pragma unroll
    for (int j = 0; j < 4; ++j) acc[i][j] = zero_v8f();

  const int arow = t >> 1;
  const int acb  = (t & 1) << 4;
  const int bk   = t >> 3;
  const int bnb  = (t & 7) << 4;

  for (int k0 = 0; k0 < EMB_; k0 += 32) {
    __syncthreads();
    // stage A (already bf16): two 16-byte loads per thread
    {
      const __bf16* src = ctx + (size_t)(gm0 + arow) * EMB_ + k0 + acb;
      if (k0 + 32 < EMB_) __builtin_prefetch(src + 32, 0, 1);
      bf16x8 v0 = *reinterpret_cast<const bf16x8*>(src);
      bf16x8 v1 = *reinterpret_cast<const bf16x8*>(src + 8);
      *reinterpret_cast<bf16x8*>(&As[arow][acb])     = v0;
      *reinterpret_cast<bf16x8*>(&As[arow][acb + 8]) = v1;
    }
    // stage B transposed (f32 -> bf16)
    {
      const float* src = Wo + (size_t)(k0 + bk) * EMB_ + n0 + bnb;
#pragma unroll
      for (int i = 0; i < 16; i += 4) {
        float4 v = *reinterpret_cast<const float4*>(src + i);
        Bs[bnb + i + 0][bk] = (__bf16)v.x;
        Bs[bnb + i + 1][bk] = (__bf16)v.y;
        Bs[bnb + i + 2][bk] = (__bf16)v.z;
        Bs[bnb + i + 3][bk] = (__bf16)v.w;
      }
    }
    __syncthreads();

    v16bf a0 = load_frag_a(&As[wm * 32 +  0][0], LDSTR, lane);
    v16bf a1 = load_frag_a(&As[wm * 32 + 16][0], LDSTR, lane);
#pragma unroll
    for (int bi = 0; bi < 4; ++bi) {
      v16bf b = load_frag_b(&Bs[wn * 64 + bi * 16][0], LDSTR, lane);
      acc[0][bi] = wmma_bf16(a0, b, acc[0][bi]);
      acc[1][bi] = wmma_bf16(a1, b, acc[1][bi]);
    }
  }

  const int nlane = lane & 15;
  const int mgrp  = (lane >> 4) << 3;
#pragma unroll
  for (int ai = 0; ai < 2; ++ai) {
#pragma unroll
    for (int bi = 0; bi < 4; ++bi) {
      const int gn = n0 + wn * 64 + bi * 16 + nlane;
      const float bvv = bo[gn];
#pragma unroll
      for (int r = 0; r < 8; ++r) {
        const int gm = gm0 + wm * 32 + ai * 16 + mgrp + r;
        out[(size_t)gm * EMB_ + gn] = acc[ai][bi][r] + bvv;
      }
    }
  }
}

// ============================================================================
extern "C" void kernel_launch(void* const* d_in, const int* in_sizes, int n_in,
                              void* d_out, int out_size, void* d_ws, size_t ws_size,
                              hipStream_t stream) {
  const float* q  = (const float*)d_in[0];
  const float* k  = (const float*)d_in[1];
  const float* v  = (const float*)d_in[2];
  const float* Wq = (const float*)d_in[3];
  const float* bq = (const float*)d_in[4];
  const float* Wk = (const float*)d_in[5];
  const float* bk = (const float*)d_in[6];
  const float* Wv = (const float*)d_in[7];
  const float* bv = (const float*)d_in[8];
  const float* Wo = (const float*)d_in[9];
  const float* bo = (const float*)d_in[10];
  float* out = (float*)d_out;

  const size_t HSZ = (size_t)B_ * HEADS_ * S_ * DEPTH_;   // 8M elems
  __bf16* qh  = (__bf16*)d_ws;
  __bf16* kh  = qh  + HSZ;
  __bf16* vT  = kh  + HSZ;
  __bf16* ctx = vT  + HSZ;                                 // total 64 MiB bf16

  const dim3 blk(256);
  proj_kernel<false><<<dim3(MROWS_ / 128, HEADS_ / 2), blk, 0, stream>>>(q, Wq, bq, qh);
  proj_kernel<false><<<dim3(MROWS_ / 128, HEADS_ / 2), blk, 0, stream>>>(k, Wk, bk, kh);
  proj_kernel<true ><<<dim3(MROWS_ / 128, HEADS_ / 2), blk, 0, stream>>>(v, Wv, bv, vT);
  attn_kernel      <<<dim3(S_ / 128, B_ * HEADS_),     blk, 0, stream>>>(qh, kh, vT, ctx);
  outproj_kernel   <<<dim3(MROWS_ / 128, EMB_ / 128),  blk, 0, stream>>>(ctx, Wo, bo, out);
}
